// DCGAN_C_55396488184571
// MI455X (gfx1250) — matrix operand order repaired and checked by reading, not compiled, not run on hardware
//
#include <hip/hip_runtime.h>
#include <hip/hip_bf16.h>

// ---------------------------------------------------------------------------
// DCGAN discriminator + relation network, MI455X (gfx1250, wave32, WMMA).
// Layer-1 of the pair MLP is factorized (U[i] + V[j]); layers 2-4 run as
// fused f16 WMMA GEMMs: activations AND per-layer weights resident in LDS
// (weights staged once per workgroup per layer -> 8x less L2 traffic).
// ---------------------------------------------------------------------------

typedef __attribute__((ext_vector_type(16))) _Float16 v16h;
typedef __attribute__((ext_vector_type(8)))  _Float16 v8h;
typedef __attribute__((ext_vector_type(8)))  float    v8f;

__device__ __forceinline__ v16h make_frag(v8h lo, v8h hi) {
  v16h r;
#pragma unroll
  for (int t = 0; t < 8; ++t) { r[t] = lo[t]; r[t + 8] = hi[t]; }
  return r;
}

__device__ __forceinline__ float lrelu(float x) { return x > 0.f ? x : 0.2f * x; }

// ------------------------- conv backbone (VALU) ----------------------------

// conv1: [32,12,64,64] -> [32,64,32,32], 4x4 s2 p1, + leaky relu
__global__ __launch_bounds__(256) void conv1_k(const float* __restrict__ img,
                                               const float* __restrict__ w,
                                               float* __restrict__ out) {
  int idx = blockIdx.x * 256 + threadIdx.x;             // 2^21 total
  int ox = idx & 31, oy = (idx >> 5) & 31, oc = (idx >> 10) & 63, b = idx >> 16;
  float s = 0.f;
  for (int ic = 0; ic < 12; ++ic) {
    const float* ip = img + ((b * 12 + ic) * 64) * 64;
    const float* wp = w + ((oc * 12 + ic) * 4) * 4;
#pragma unroll
    for (int ky = 0; ky < 4; ++ky) {
      int iy = oy * 2 - 1 + ky;
      if ((unsigned)iy >= 64u) continue;
#pragma unroll
      for (int kx = 0; kx < 4; ++kx) {
        int ix = ox * 2 - 1 + kx;
        if ((unsigned)ix >= 64u) continue;
        s += ip[iy * 64 + ix] * wp[ky * 4 + kx];
      }
    }
  }
  out[idx] = lrelu(s);
}

// conv2: [32,64,32,32] -> raw [32,128,16,16]; block=(b,oc), fused BN stats
__global__ __launch_bounds__(256) void conv2_k(const float* __restrict__ in,
                                               const float* __restrict__ w,
                                               float* __restrict__ outRaw,
                                               float* __restrict__ sum,
                                               float* __restrict__ sq) {
  __shared__ float r1[256], r2[256];
  int b = blockIdx.x >> 7, oc = blockIdx.x & 127;
  int tid = threadIdx.x;
  int oy = tid >> 4, ox = tid & 15;
  float s = 0.f;
  for (int ic = 0; ic < 64; ++ic) {
    const float* ip = in + ((b * 64 + ic) * 32) * 32;
    const float* wp = w + ((oc * 64 + ic) * 4) * 4;
#pragma unroll
    for (int ky = 0; ky < 4; ++ky) {
      int iy = oy * 2 - 1 + ky;
      if ((unsigned)iy >= 32u) continue;
#pragma unroll
      for (int kx = 0; kx < 4; ++kx) {
        int ix = ox * 2 - 1 + kx;
        if ((unsigned)ix >= 32u) continue;
        s += ip[iy * 32 + ix] * wp[ky * 4 + kx];
      }
    }
  }
  outRaw[((b * 128 + oc) * 16 + oy) * 16 + ox] = s;
  r1[tid] = s; r2[tid] = s * s;
  __syncthreads();
  for (int st = 128; st > 0; st >>= 1) {
    if (tid < st) { r1[tid] += r1[tid + st]; r2[tid] += r2[tid + st]; }
    __syncthreads();
  }
  if (tid == 0) { atomicAdd(&sum[oc], r1[0]); atomicAdd(&sq[oc], r2[0]); }
}

__global__ __launch_bounds__(256) void bn2_k(const float* __restrict__ raw,
                                             const float* __restrict__ sum,
                                             const float* __restrict__ sq,
                                             const float* __restrict__ g,
                                             const float* __restrict__ bb,
                                             float* __restrict__ out) {
  int b = blockIdx.x >> 7, oc = blockIdx.x & 127, tid = threadIdx.x;
  float m = sum[oc] * (1.f / 8192.f);
  float var = sq[oc] * (1.f / 8192.f) - m * m;
  float sc = g[oc] * rsqrtf(var + 1e-5f);
  float sh = bb[oc] - m * sc;
  int idx = (b * 128 + oc) * 256 + tid;
  out[idx] = lrelu(raw[idx] * sc + sh);
}

// conv3: [32,128,16,16] -> raw [32,256,8,8]; block=(b,oc), 64 threads
__global__ __launch_bounds__(64) void conv3_k(const float* __restrict__ in,
                                              const float* __restrict__ w,
                                              float* __restrict__ outRaw,
                                              float* __restrict__ sum,
                                              float* __restrict__ sq) {
  __shared__ float r1[64], r2[64];
  int b = blockIdx.x >> 8, oc = blockIdx.x & 255;
  int tid = threadIdx.x;
  int oy = tid >> 3, ox = tid & 7;
  float s = 0.f;
  for (int ic = 0; ic < 128; ++ic) {
    const float* ip = in + ((b * 128 + ic) * 16) * 16;
    const float* wp = w + ((oc * 128 + ic) * 4) * 4;
#pragma unroll
    for (int ky = 0; ky < 4; ++ky) {
      int iy = oy * 2 - 1 + ky;
      if ((unsigned)iy >= 16u) continue;
#pragma unroll
      for (int kx = 0; kx < 4; ++kx) {
        int ix = ox * 2 - 1 + kx;
        if ((unsigned)ix >= 16u) continue;
        s += ip[iy * 16 + ix] * wp[ky * 4 + kx];
      }
    }
  }
  outRaw[((b * 256 + oc) * 8 + oy) * 8 + ox] = s;
  r1[tid] = s; r2[tid] = s * s;
  __syncthreads();
  for (int st = 32; st > 0; st >>= 1) {
    if (tid < st) { r1[tid] += r1[tid + st]; r2[tid] += r2[tid + st]; }
    __syncthreads();
  }
  if (tid == 0) { atomicAdd(&sum[oc], r1[0]); atomicAdd(&sq[oc], r2[0]); }
}

__global__ __launch_bounds__(64) void bn3_k(const float* __restrict__ raw,
                                            const float* __restrict__ sum,
                                            const float* __restrict__ sq,
                                            const float* __restrict__ g,
                                            const float* __restrict__ bb,
                                            float* __restrict__ out) {
  int b = blockIdx.x >> 8, oc = blockIdx.x & 255, tid = threadIdx.x;
  float m = sum[oc] * (1.f / 2048.f);
  float var = sq[oc] * (1.f / 2048.f) - m * m;
  float sc = g[oc] * rsqrtf(var + 1e-5f);
  float sh = bb[oc] - m * sc;
  int idx = (b * 256 + oc) * 64 + tid;
  out[idx] = lrelu(raw[idx] * sc + sh);
}

// ----------------- feature build + weight repack (f16) ---------------------

// X  [2048][288]: [enc(256), coord(1), pad]   (row = b*64 + n)
// XA [2048][288]: [enc(256), aux(10), coord(1), pad]
__global__ void build_features(const float* __restrict__ h3,
                               const float* __restrict__ aux,
                               _Float16* __restrict__ X,
                               _Float16* __restrict__ XA) {
  int idx = blockIdx.x * 256 + threadIdx.x;
  if (idx >= 2048 * 288) return;
  int row = idx / 288, k = idx - row * 288;
  int b = row >> 6, n = row & 63;
  float xv = 0.f, xav = 0.f;
  if (k < 256) {
    float v = h3[(b * 256 + k) * 64 + n];   // enc[b,n,l] = h3[b,l,n]
    xv = v; xav = v;
  } else if (k == 256) {
    xv = (float)n; xav = aux[b * 10 + 0];
  } else if (k < 266) {
    xav = aux[b * 10 + (k - 256)];
  } else if (k == 266) {
    xav = (float)n;
  }
  X[idx] = (_Float16)xv;
  XA[idx] = (_Float16)xav;
}

// Repack W [Ksrc x 256] (row-major, K = reduction dim) into chunked WMMA
// B-fragment order: [ntile(16)][kc][chunk(2)][lane(32)][q(8)].
// Lane l covers K-slice kc*32 + chunk*16 + (l>=16 ? 8:0) + q, column
// n = ntile*16 + (l&15).  Chunked so each 16B ds/global load has 16B lane
// stride: conflict-free LDS bank quads, fully contiguous global reads.
__global__ void repack_w(const float* __restrict__ W, _Float16* __restrict__ out,
                         int Ksrc, int kchunks) {
  int idx = blockIdx.x * 256 + threadIdx.x;
  int total = 16 * kchunks * 512;
  if (idx >= total) return;
  int q    = idx & 7;
  int lane = (idx >> 3) & 31;
  int c    = (idx >> 8) & 1;
  int kc   = (idx >> 9) % kchunks;
  int nt   = idx / (kchunks * 512);
  int k = kc * 32 + c * 16 + (lane >> 4) * 8 + q;
  int n = nt * 16 + (lane & 15);
  out[idx] = (k < Ksrc) ? (_Float16)W[k * 256 + n] : (_Float16)0.f;
}

// ------------------- layer-1 factors: U = X@W1a, V = XA@W1b ----------------

__global__ __launch_bounds__(256) void gemm_uv(const _Float16* __restrict__ X,
                                               const _Float16* __restrict__ Wsw,
                                               float* __restrict__ O) {
  const int tid = threadIdx.x;
  const int lane = tid & 31;
  const int wave = tid >> 5;
  const int tile = blockIdx.x * 8 + wave;   // 128 row-tiles * 16 col-tiles
  const int rt = tile >> 4;
  const int nt = tile & 15;
  const int lrow = lane & 15;
  const int hi = lane >> 4;
  v8f acc = {};
#pragma unroll
  for (int kc = 0; kc < 9; ++kc) {
    const _Float16* ap = X + (rt * 16 + lrow) * 288 + kc * 32 + hi * 8;
    v16h a = make_frag(*(const v8h*)ap, *(const v8h*)(ap + 16));
    const _Float16* bp = Wsw + ((nt * 9 + kc) * 64 + lane) * 8;
    v16h bm = make_frag(*(const v8h*)bp, *(const v8h*)(bp + 256));
    acc = __builtin_amdgcn_wmma_f32_16x16x32_f16(false, a, false, bm,
                                                 (short)0, acc, false, false);
  }
#pragma unroll
  for (int v = 0; v < 8; ++v)
    O[(rt * 16 + v + hi * 8) * 256 + nt * 16 + lrow] = acc[v];
}

// -------- fused pair MLP: relu(U[i]+V[j]+b1) -> W2 -> W3 -> W4 -> sum -------

#define ACT_STRIDE 264   // 256 + 8 halves pad: 528B row stride, conflict-free
#define WHALVES    (16 * 8 * 512)   // 65536 halves = 128KB per weight matrix

__global__ __launch_bounds__(256) void rn_mlp_fused(
    const float* __restrict__ U, const float* __restrict__ V,
    const float* __restrict__ gb1,
    const _Float16* __restrict__ W2sw, const float* __restrict__ gb2,
    const _Float16* __restrict__ W3sw, const float* __restrict__ gb3,
    const _Float16* __restrict__ W4sw, const float* __restrict__ gb4,
    float* __restrict__ pooled) {
  extern __shared__ char smem_raw[];
  _Float16* wbuf = (_Float16*)smem_raw;            // 65536 halves (128KB)
  _Float16* bufA = wbuf + WHALVES;                 // 128 x 264 halves
  _Float16* bufB = bufA + 128 * ACT_STRIDE;

  const int tid = threadIdx.x;
  const int b = blockIdx.y;
  const int rowBase = blockIdx.x * 128;            // 128 of the 4096 pair rows

  // stage layer-1 activations: act0[r][k] = relu(U[b,i,k] + V[b,j,k] + gb1[k])
  for (int idx = tid; idx < 128 * 256; idx += 256) {
    int r = idx >> 8, k = idx & 255;
    int p = rowBase + r;
    int i = p & 63, j = p >> 6;                    // pair (j,i): x from i, xa from j
    float val = U[(b * 64 + i) * 256 + k] + V[(b * 64 + j) * 256 + k] + gb1[k];
    bufA[r * ACT_STRIDE + k] = (_Float16)fmaxf(val, 0.f);
  }

  const int lane = tid & 31;
  const int wave = tid >> 5;
  const int m0 = wave * 16;                        // each wave owns 16 rows
  const int lrow = lane & 15;
  const int hi = lane >> 4;

  const _Float16* Wl[3] = {W2sw, W3sw, W4sw};
  const float* bl[3] = {gb2, gb3, gb4};

#pragma unroll 1
  for (int l = 0; l < 3; ++l) {
    // cooperatively stage this layer's 128KB weight matrix into LDS
    {
      const uint4* src4 = (const uint4*)Wl[l];
      uint4* dst4 = (uint4*)wbuf;
#pragma unroll
      for (int t = 0; t < 32; ++t)                 // 8192 uint4 / 256 threads
        dst4[t * 256 + tid] = src4[t * 256 + tid];
    }
    __syncthreads();                               // wbuf ready; act also ready

    const _Float16* src = (l & 1) ? bufB : bufA;
    _Float16* dst = (l & 1) ? bufA : bufB;
    const float* bias = bl[l];
    for (int nt = 0; nt < 16; ++nt) {
      const int col = nt * 16 + lrow;
      float bv = bias[col];
      v8f acc = {bv, bv, bv, bv, bv, bv, bv, bv};
#pragma unroll
      for (int kc = 0; kc < 8; ++kc) {
        const _Float16* ap = src + (m0 + lrow) * ACT_STRIDE + kc * 32 + hi * 8;
        v16h a = make_frag(*(const v8h*)ap, *(const v8h*)(ap + 16));
        const _Float16* bp = wbuf + ((nt * 8 + kc) * 64 + lane) * 8;
        v16h bm = make_frag(*(const v8h*)bp, *(const v8h*)(bp + 256));
        acc = __builtin_amdgcn_wmma_f32_16x16x32_f16(false, a, false, bm,
                                                     (short)0, acc, false, false);
      }
      if (l < 2) {
#pragma unroll
        for (int v = 0; v < 8; ++v) {
          int row = m0 + v + hi * 8;
          dst[row * ACT_STRIDE + col] = (_Float16)fmaxf(acc[v], 0.f);
        }
      } else {
        float s = 0.f;
#pragma unroll
        for (int v = 0; v < 8; ++v) s += fmaxf(acc[v], 0.f);
        s += __shfl_xor(s, 16, 32);                // merge complementary rows
        if (hi == 0) atomicAdd(&pooled[b * 256 + col], s);
      }
    }
    __syncthreads();   // act dst complete; safe to overwrite wbuf next layer
  }
}

// --------------------------- final f-MLP -----------------------------------

__global__ __launch_bounds__(256) void fmlp_k(const float* __restrict__ pooled,
                                              const float* __restrict__ fW1,
                                              const float* __restrict__ fb1,
                                              const float* __restrict__ fW2,
                                              const float* __restrict__ fb2,
                                              const float* __restrict__ fW3,
                                              const float* __restrict__ fb3,
                                              float* __restrict__ out) {
  __shared__ float p[256], h[256], r[256];
  int b = blockIdx.x, tid = threadIdx.x;
  p[tid] = pooled[b * 256 + tid];
  __syncthreads();
  float s = fb1[tid];
  for (int k = 0; k < 256; ++k) s += p[k] * fW1[k * 256 + tid];
  h[tid] = fmaxf(s, 0.f);
  __syncthreads();
  s = fb2[tid];
  for (int k = 0; k < 256; ++k) s += h[k] * fW2[k * 256 + tid];
  r[tid] = fmaxf(s, 0.f);
  __syncthreads();
  p[tid] = r[tid] * fW3[tid];
  __syncthreads();
  for (int st = 128; st > 0; st >>= 1) {
    if (tid < st) p[tid] += p[tid + st];
    __syncthreads();
  }
  if (tid == 0) out[b] = p[0] + fb3[0];
}

// ------------------------------ launcher -----------------------------------

extern "C" void kernel_launch(void* const* d_in, const int* in_sizes, int n_in,
                              void* d_out, int out_size, void* d_ws, size_t ws_size,
                              hipStream_t stream) {
  (void)in_sizes; (void)n_in; (void)out_size; (void)ws_size;
  const float* img = (const float*)d_in[0];
  const float* aux = (const float*)d_in[1];
  const float* w1  = (const float*)d_in[2];
  const float* w2  = (const float*)d_in[3];
  const float* g2  = (const float*)d_in[4];
  const float* b2  = (const float*)d_in[5];
  const float* w3c = (const float*)d_in[6];
  const float* g3  = (const float*)d_in[7];
  const float* b3  = (const float*)d_in[8];
  const float* gW1 = (const float*)d_in[9];
  const float* gb1 = (const float*)d_in[10];
  const float* gW2 = (const float*)d_in[11];
  const float* gb2 = (const float*)d_in[12];
  const float* gW3 = (const float*)d_in[13];
  const float* gb3 = (const float*)d_in[14];
  const float* gW4 = (const float*)d_in[15];
  const float* gb4 = (const float*)d_in[16];
  const float* fW1 = (const float*)d_in[17];
  const float* fb1 = (const float*)d_in[18];
  const float* fW2 = (const float*)d_in[19];
  const float* fb2 = (const float*)d_in[20];
  const float* fW3 = (const float*)d_in[21];
  const float* fb3 = (const float*)d_in[22];
  float* out = (float*)d_out;

  char* ws = (char*)d_ws;
  size_t off = 0;
  auto alloc = [&](size_t bytes) -> char* {
    char* p = ws + off;
    off += (bytes + 255) & ~(size_t)255;
    return p;
  };
  float*    h1     = (float*)alloc(32ull * 64 * 32 * 32 * 4);
  float*    h2raw  = (float*)alloc(32ull * 128 * 16 * 16 * 4);
  float*    h2     = (float*)alloc(32ull * 128 * 16 * 16 * 4);
  float*    h3raw  = (float*)alloc(32ull * 256 * 8 * 8 * 4);
  float*    h3     = (float*)alloc(32ull * 256 * 8 * 8 * 4);
  float*    bnS    = (float*)alloc(768 * 4);   // [sum2:128][sq2:128][sum3:256][sq3:256]
  _Float16* X      = (_Float16*)alloc(2048ull * 288 * 2);
  _Float16* XA     = (_Float16*)alloc(2048ull * 288 * 2);
  _Float16* W1a    = (_Float16*)alloc(16ull * 9 * 512 * 2);
  _Float16* W1b    = (_Float16*)alloc(16ull * 9 * 512 * 2);
  _Float16* W2s    = (_Float16*)alloc(16ull * 8 * 512 * 2);
  _Float16* W3s    = (_Float16*)alloc(16ull * 8 * 512 * 2);
  _Float16* W4s    = (_Float16*)alloc(16ull * 8 * 512 * 2);
  float*    Um     = (float*)alloc(2048ull * 256 * 4);
  float*    Vm     = (float*)alloc(2048ull * 256 * 4);
  float*    pooled = (float*)alloc(32ull * 256 * 4);

  hipMemsetAsync(bnS, 0, 768 * 4, stream);
  hipMemsetAsync(pooled, 0, 32 * 256 * 4, stream);

  // conv backbone
  conv1_k<<<(32 * 64 * 32 * 32) / 256, 256, 0, stream>>>(img, w1, h1);
  conv2_k<<<32 * 128, 256, 0, stream>>>(h1, w2, h2raw, bnS, bnS + 128);
  bn2_k<<<32 * 128, 256, 0, stream>>>(h2raw, bnS, bnS + 128, g2, b2, h2);
  conv3_k<<<32 * 256, 64, 0, stream>>>(h2, w3c, h3raw, bnS + 256, bnS + 512);
  bn3_k<<<32 * 256, 64, 0, stream>>>(h3raw, bnS + 256, bnS + 512, g3, b3, h3);

  // features + weight repacks (f16, chunked WMMA B-fragment layout)
  build_features<<<(2048 * 288 + 255) / 256, 256, 0, stream>>>(h3, aux, X, XA);
  repack_w<<<(16 * 9 * 512 + 255) / 256, 256, 0, stream>>>(gW1, W1a, 257, 9);
  repack_w<<<(16 * 9 * 512 + 255) / 256, 256, 0, stream>>>(gW1 + 257 * 256, W1b, 267, 9);
  repack_w<<<(16 * 8 * 512 + 255) / 256, 256, 0, stream>>>(gW2, W2s, 256, 8);
  repack_w<<<(16 * 8 * 512 + 255) / 256, 256, 0, stream>>>(gW3, W3s, 256, 8);
  repack_w<<<(16 * 8 * 512 + 255) / 256, 256, 0, stream>>>(gW4, W4s, 256, 8);

  // factorized layer-1 halves
  gemm_uv<<<256, 256, 0, stream>>>(X, W1a, Um);
  gemm_uv<<<256, 256, 0, stream>>>(XA, W1b, Vm);

  // fused pair-MLP (layers 2-4) + sum pooling
  dim3 grid(32, 32);  // 32 row-chunks of 128 pairs x 32 batch
  size_t shmem = (size_t)(WHALVES + 2 * 128 * ACT_STRIDE) * sizeof(_Float16);
  (void)hipFuncSetAttribute((const void*)rn_mlp_fused,
                            hipFuncAttributeMaxDynamicSharedMemorySize,
                            (int)shmem);
  rn_mlp_fused<<<grid, 256, shmem, stream>>>(Um, Vm, gb1, W2s, gb2, W3s, gb3,
                                             W4s, gb4, pooled);

  // final f MLP
  fmlp_k<<<32, 256, 0, stream>>>(pooled, fW1, fb1, fW2, fb2, fW3, fb3, out);
}
